// decoder_22093311771013
// MI455X (gfx1250) — matrix operand order repaired and checked by reading, not compiled, not run on hardware
//
#include <hip/hip_runtime.h>
#include <hip/hip_bf16.h>
#include <math.h>

// ---------------------------------------------------------------------------
// Problem constants (from reference)
// ---------------------------------------------------------------------------
#define BB   64      // batch
#define QQ   64      // channels
#define LL   8192    // spectral length
#define KC1  51      // conv1 taps (symmetric, shared across channels)
#define KC2  501     // conv2 taps (symmetric, per channel)
#define HC1  26
#define HT2  251

// WMMA tiling: each 16x16 D tile covers 256 consecutive outputs:
//   out[lbase + 16*n + m] = D[m][n]
// conv1: band width 51+15 -> 3 K-chunks of 32 ; conv2: 501+15 -> 17 chunks.
#define NCH1 3
#define NCH2 17

// A-fragment table: per chunk, [lane(32)][vgpr(8)] packed 2xbf16 dwords = 1 KB.
#define FRAG_DW_PER_CHUNK 256              // 32 lanes * 8 dwords
#define A1_DWORDS (NCH1 * FRAG_DW_PER_CHUNK)          // 768
#define A2_DWORDS_PER_Q (NCH2 * FRAG_DW_PER_CHUNK)    // 4352

// LDS padded rows (bf16 elements). conv1 window origin = -26 (keeps bases
// 8-element aligned => 16B-aligned ds_load_b128); conv2 origin = -250.
#define XPAD_ELEMS 8288   // valid x at [26, 26+8192)
#define SPAD_ELEMS 8976   // valid s at [250, 250+8192)

// dynamic-LDS layout (bytes), all offsets 16B aligned
#define SM_AFR   0                         // 5120 dwords = 20480 B (A1 then A2)
#define SM_XPAD  20480                     // 8288 * 2 = 16576 B
#define SM_SPAD  37056                     // 8976 * 2 = 17952 B
#define SM_OUT   55008                     // 8192 * 4 = 32768 B
#define SM_RED   87776                     // 16 floats
#define SM_TOTAL 87840

// workspace layout (bytes). Requires ws_size >= ~136.3 MB (y buffer).
#define WS_A1    0
#define WS_A2    4096
#define WS_A     1118208    // 4096 + 64*4352*4 = 1,118,208 ; a[64] f32
#define WS_INV   1118464    // inv_norm scalar
#define WS_Y     2097152    // y[B][Q][L] f32 = 128 MB

typedef __attribute__((ext_vector_type(16))) __bf16 v16bf;
typedef __attribute__((ext_vector_type(8)))  float  v8f;
typedef __attribute__((ext_vector_type(8)))  int    vi8;

static __device__ __forceinline__ unsigned short f2bf(float f) {
  unsigned u = __builtin_bit_cast(unsigned, f);
  u += 0x7FFFu + ((u >> 16) & 1u);          // round-to-nearest-even
  return (unsigned short)(u >> 16);
}

static __device__ __forceinline__ float sigm(float x) {
  return 1.0f / (1.0f + expf(-x));
}

// Load a 32-byte WMMA operand fragment (works for LDS or global pointers;
// 16B-aligned by construction -> two b128 loads).
static __device__ __forceinline__ v16bf frag_ld(const void* p) {
  const int4* q = (const int4*)p;
  int4 lo = q[0];
  int4 hi = q[1];
  vi8 t;
  t[0] = lo.x; t[1] = lo.y; t[2] = lo.z; t[3] = lo.w;
  t[4] = hi.x; t[5] = hi.y; t[6] = hi.z; t[7] = hi.w;
  return __builtin_bit_cast(v16bf, t);
}

// ---------------------------------------------------------------------------
// Kernel 0: build symmetric kernels, sigmoid params, and pre-swizzled WMMA
// A-fragments (banded-Toeplitz weight slices in this code's contiguous-K A-matrix VGPR layout:
// lane = m + 16*half ; vgpr v holds K = 2v + 16*half, +1).
// grid = 64 blocks (one per channel q), 256 threads.
// ---------------------------------------------------------------------------
__global__ __launch_bounds__(256) void prep_kernel(
    const float* __restrict__ cw,     // conv_w_half [26]
    const float* __restrict__ tw,     // tc_w_half   [64][251]
    const float* __restrict__ araw,   // A_raw       [64]
    unsigned int* __restrict__ afrA1, // [3][32][8]  dwords
    unsigned int* __restrict__ afrA2, // [64][17][32][8] dwords
    float* __restrict__ aOut,         // [64]
    float* __restrict__ invOut)       // [1]
{
  __shared__ float kt[KC2];   // 501-tap symmetric sigmoid kernel for this q
  __shared__ float kc[64];    // 51-tap symmetric shared kernel (block 0 only)

  const int q = blockIdx.x;
  const int i = threadIdx.x;

  if (i < HT2) {
    float s = sigm(tw[q * HT2 + i]);
    kt[i] = s;
    if (i < HT2 - 1) kt[KC2 - 1 - i] = s;   // symmetric: kt[k]=kt[500-k]
  }
  __syncthreads();

  const int lane = i & 31;
  const int v    = i >> 5;                   // 0..7
  const int m    = lane & 15;
  const int k0   = 2 * v + 16 * (lane >> 4);

  unsigned int* dst = afrA2 + (size_t)q * A2_DWORDS_PER_Q;
#pragma unroll
  for (int p = 0; p < NCH2; ++p) {
    int c  = 32 * p + k0;
    int i0 = c - m;                          // A2[m][c] = kt[c-m]
    float f0 = (i0 >= 0 && i0 < KC2)     ? kt[i0]     : 0.0f;
    float f1 = (i0 + 1 >= 0 && i0 + 1 < KC2) ? kt[i0 + 1] : 0.0f;
    dst[(p * 32 + lane) * 8 + v] =
        (unsigned)f2bf(f0) | ((unsigned)f2bf(f1) << 16);
  }

  if (q == 0) {
    if (i < HC1) {
      float s = cw[i];
      kc[i] = s;
      if (i < HC1 - 1) kc[KC1 - 1 - i] = s;
    }
    __syncthreads();                         // block-uniform, legal
#pragma unroll
    for (int p = 0; p < NCH1; ++p) {
      int c  = 32 * p + k0;
      int i0 = c - m - 1;                    // A1[m][c] = kc[c-m-1] (origin -26)
      float f0 = (i0 >= 0 && i0 < KC1)     ? kc[i0]     : 0.0f;
      float f1 = (i0 + 1 >= 0 && i0 + 1 < KC1) ? kc[i0 + 1] : 0.0f;
      afrA1[(p * 32 + lane) * 8 + v] =
          (unsigned)f2bf(f0) | ((unsigned)f2bf(f1) << 16);
    }
    if (i == 0) {
      float prod = 1.0f;
      for (int qq = 0; qq < QQ; ++qq) {
        float aa = sigm(araw[qq]);
        aOut[qq] = aa;
        prod *= aa;
      }
      float peak = sigm(tw[(QQ - 1) * HT2 + (HT2 - 1)]);  // wt[-1,-1]
      invOut[0] = 1.0f / (peak * prod);
    }
  }
}

// ---------------------------------------------------------------------------
// Kernel 1: fully fused per-(b,q) row: conv1(WMMA bf16) -> softmax ->
// conv2(WMMA bf16) -> y. grid = B*Q = 4096 workgroups, 256 threads (8 waves),
// wave w owns output tiles {w, w+8, w+16, w+24} (4 tiles x 256 outputs).
// ---------------------------------------------------------------------------
__global__ __launch_bounds__(256) void fused_row_kernel(
    const float* __restrict__ x,
    const unsigned int* __restrict__ afrA1,
    const unsigned int* __restrict__ afrA2,
    float* __restrict__ y)
{
  extern __shared__ char smem[];
  unsigned int*   sAfr = (unsigned int*)  (smem + SM_AFR);   // 5120 dwords
  unsigned short* sXp  = (unsigned short*)(smem + SM_XPAD);  // bf16 x row (pad 26)
  unsigned short* sSp  = (unsigned short*)(smem + SM_SPAD);  // bf16 softmax row (pad 250)
  float*          sOut = (float*)         (smem + SM_OUT);   // f32 conv2 row
  float*          sRed = (float*)         (smem + SM_RED);   // reductions

  const int tid = threadIdx.x;
  const int wg  = blockIdx.x;
  const int q   = wg & 63;

  // ---- stage A fragments into LDS (A1 then this q's A2) -------------------
  for (int i = tid; i < A1_DWORDS; i += 256) sAfr[i] = afrA1[i];
  const unsigned int* a2g = afrA2 + (size_t)q * A2_DWORDS_PER_Q;
  for (int i = tid; i < A2_DWORDS_PER_Q; i += 256) sAfr[A1_DWORDS + i] = a2g[i];

  // ---- zero the padded rows ----------------------------------------------
  unsigned int* xp32 = (unsigned int*)sXp;
  unsigned int* sp32 = (unsigned int*)sSp;
  for (int i = tid; i < XPAD_ELEMS / 2; i += 256) xp32[i] = 0u;
  for (int i = tid; i < SPAD_ELEMS / 2; i += 256) sp32[i] = 0u;
  __syncthreads();

  // ---- load x row, convert to bf16 into xpad (x index e-26) ---------------
  const float* xr = x + (size_t)wg * LL;
#pragma unroll
  for (int it = 0; it < 16; ++it) {
    int e = it * 512 + tid * 2;
    float2 v2 = *(const float2*)(xr + e);
    xp32[(26 + e) >> 1] =
        (unsigned)f2bf(v2.x) | ((unsigned)f2bf(v2.y) << 16);
  }
  __syncthreads();

  const int wv = tid >> 5;   // wave 0..7
  const int t  = tid & 31;   // lane
  const int n  = t & 15;     // D column / B column
  const int h  = t >> 4;     // lane half

  // ---- conv1 via WMMA: acc[tile][m-row] holds 256 outputs per tile --------
  v8f acc[4];
#pragma unroll
  for (int ti = 0; ti < 4; ++ti) {
    int lbase = (wv + ti * 8) << 8;
    v8f c;
#pragma unroll
    for (int r = 0; r < 8; ++r) c[r] = 0.0f;
#pragma unroll
    for (int p = 0; p < NCH1; ++p) {
      v16bf A = frag_ld(&sAfr[(p * 32 + t) * 8]);
      int base = lbase + 16 * n + 32 * p + 16 * h;        // 8-elem aligned
      v16bf B = frag_ld(&sXp[base]);
      c = __builtin_amdgcn_wmma_f32_16x16x32_bf16(
              false, A, false, B, (short)0, c, false, false);
    }
    acc[ti] = c;
  }

  // ---- softmax over the row (values live in registers) --------------------
  float mx = -__builtin_inff();
#pragma unroll
  for (int ti = 0; ti < 4; ++ti)
#pragma unroll
    for (int r = 0; r < 8; ++r) mx = fmaxf(mx, acc[ti][r]);
#pragma unroll
  for (int o = 16; o > 0; o >>= 1) mx = fmaxf(mx, __shfl_xor(mx, o, 32));
  if (t == 0) sRed[wv] = mx;
  __syncthreads();
  float bm = sRed[0];
#pragma unroll
  for (int k = 1; k < 8; ++k) bm = fmaxf(bm, sRed[k]);

  float sum = 0.0f;
#pragma unroll
  for (int ti = 0; ti < 4; ++ti)
#pragma unroll
    for (int r = 0; r < 8; ++r) {
      float e = expf(acc[ti][r] - bm);
      acc[ti][r] = e;
      sum += e;
    }
#pragma unroll
  for (int o = 16; o > 0; o >>= 1) sum += __shfl_xor(sum, o, 32);
  if (t == 0) sRed[8 + wv] = sum;
  __syncthreads();
  float tot = 0.0f;
#pragma unroll
  for (int k = 0; k < 8; ++k) tot += sRed[8 + k];
  float inv = 1.0f / tot;

  // scatter softmax values (bf16) into padded spad; D elem (ti,r,lane) is
  // output index lbase + 16n + r + 8h
#pragma unroll
  for (int ti = 0; ti < 4; ++ti) {
    int lbase = (wv + ti * 8) << 8;
#pragma unroll
    for (int r = 0; r < 8; ++r)
      sSp[250 + lbase + 16 * n + r + 8 * h] = f2bf(acc[ti][r] * inv);
  }
  __syncthreads();

  // ---- conv2 via WMMA (501 taps -> 17 K-chunks) ---------------------------
#pragma unroll
  for (int ti = 0; ti < 4; ++ti) {
    int lbase = (wv + ti * 8) << 8;
    v8f c;
#pragma unroll
    for (int r = 0; r < 8; ++r) c[r] = 0.0f;
    for (int p = 0; p < NCH2; ++p) {
      v16bf A = frag_ld(&sAfr[((NCH1 + p) * 32 + t) * 8]);
      int base = lbase + 16 * n + 32 * p + 16 * h;
      v16bf B = frag_ld(&sSp[base]);
      c = __builtin_amdgcn_wmma_f32_16x16x32_bf16(
              false, A, false, B, (short)0, c, false, false);
    }
#pragma unroll
    for (int r = 0; r < 8; ++r)
      sOut[lbase + 16 * n + r + 8 * h] = c[r];
  }
  __syncthreads();

  // ---- coalesced row writeback (b128) -------------------------------------
  float* yr = y + (size_t)wg * LL;
  const int4* so4 = (const int4*)sOut;
  int4* yr4 = (int4*)yr;
#pragma unroll
  for (int i = tid; i < LL / 4; i += 256) yr4[i] = so4[i];
}

// ---------------------------------------------------------------------------
// Kernel 2: lattice recursion over Q per (b,l), then normalization.
// grid = B*L/256 = 2048 blocks. Coalesced 128B/wave loads per step; a[q]
// is wave-uniform -> scalar loads.
// ---------------------------------------------------------------------------
__global__ __launch_bounds__(256) void lattice_kernel(
    const float* __restrict__ y,
    const float* __restrict__ a,
    const float* __restrict__ invn,
    float* __restrict__ out)
{
  int gid = blockIdx.x * 256 + threadIdx.x;
  int b = gid >> 13;
  int l = gid & (LL - 1);
  const float* yb = y + (size_t)(b * QQ) * LL + l;

  float r = 0.0f, t2 = 1.0f;
#pragma unroll 4
  for (int q = 0; q < QQ; ++q) {
    float rn = yb[(size_t)q * LL];
    float aq = a[q];
    float tn = 1.0f - rn;
    float s  = 1.0f / (1.0f - aq * r * rn);
    float rN = r + aq * rn * t2 * s;
    t2 = aq * t2 * tn * tn * s * s;
    r = rN;
  }
  out[gid] = r * invn[0];
}

// ---------------------------------------------------------------------------
extern "C" void kernel_launch(void* const* d_in, const int* in_sizes, int n_in,
                              void* d_out, int out_size, void* d_ws, size_t ws_size,
                              hipStream_t stream) {
  const float* x    = (const float*)d_in[0];   // [64,64,8192]
  const float* cw   = (const float*)d_in[1];   // [26]
  const float* tw   = (const float*)d_in[2];   // [64,251]
  const float* araw = (const float*)d_in[3];   // [64]
  float* out = (float*)d_out;                  // [64,8192]

  char* ws = (char*)d_ws;
  unsigned int* afrA1 = (unsigned int*)(ws + WS_A1);
  unsigned int* afrA2 = (unsigned int*)(ws + WS_A2);
  float* aArr = (float*)(ws + WS_A);
  float* invN = (float*)(ws + WS_INV);
  float* yBuf = (float*)(ws + WS_Y);           // 128 MB intermediate

  prep_kernel<<<QQ, 256, 0, stream>>>(cw, tw, araw, afrA1, afrA2, aArr, invN);
  fused_row_kernel<<<BB * QQ, 256, SM_TOTAL, stream>>>(x, afrA1, afrA2, yBuf);
  lattice_kernel<<<(BB * LL) / 256, 256, 0, stream>>>(yBuf, aArr, invN, out);
  (void)in_sizes; (void)n_in; (void)out_size; (void)ws_size;
}
